// TotalLossFunction_87471303950501
// MI455X (gfx1250) — compile-verified, matrix operand order checked by reading, not run on hardware
//
#include <hip/hip_runtime.h>
#include <hip/hip_bf16.h>

typedef __attribute__((ext_vector_type(16))) __bf16 v16bf;
typedef __attribute__((ext_vector_type(8)))  float  v8f;
typedef __attribute__((ext_vector_type(4)))  unsigned int u32x4;

#define NN 8192
#define CC 1000
#define DD 128
#define BT 128
#define ALPHA_C 0.1f

__device__ __forceinline__ unsigned short f2bf(float f) {
    union { float f; unsigned u; } v; v.f = f;
    unsigned r = v.u + 0x7FFFu + ((v.u >> 16) & 1u);   // round-to-nearest-even
    return (unsigned short)(r >> 16);
}

union FragBF {
    v16bf v;
    u32x4 q[2];
};

// ---------------------------------------------------------------------------
// Fused kernel: cross = sum(L .* (Xi Xj^T))  +  row/col sums of L.
// L is read exactly once from HBM (non-temporal); X served from LDS (bf16).
// ---------------------------------------------------------------------------
__global__ __launch_bounds__(256) void cross_kernel(
    const float* __restrict__ L, const float* __restrict__ X,
    float* __restrict__ crossAcc, float* __restrict__ rowS, float* __restrict__ colS)
{
    __shared__ unsigned short sXi[BT][DD];   // 32 KB
    __shared__ unsigned short sXj[BT][DD];   // 32 KB
    __shared__ float sCol[BT];
    __shared__ float sRed[8];

    const int i0   = blockIdx.y * BT;
    const int j0   = blockIdx.x * BT;
    const int tid  = threadIdx.x;
    const int lane = tid & 31;
    const int wave = tid >> 5;
    const int h    = lane >> 4;     // half of wave
    const int ln   = lane & 15;

    // Stage X row-blocks into LDS as bf16 (float4 loads, packed stores).
    for (int idx = tid; idx < BT * (DD / 4); idx += 256) {
        int r  = idx >> 5;            // DD/4 == 32
        int c4 = (idx & 31) << 2;
        float4 fi = *(const float4*)(X + (size_t)(i0 + r) * DD + c4);
        float4 fj = *(const float4*)(X + (size_t)(j0 + r) * DD + c4);
        *(unsigned*)&sXi[r][c4 + 0] = (unsigned)f2bf(fi.x) | ((unsigned)f2bf(fi.y) << 16);
        *(unsigned*)&sXi[r][c4 + 2] = (unsigned)f2bf(fi.z) | ((unsigned)f2bf(fi.w) << 16);
        *(unsigned*)&sXj[r][c4 + 0] = (unsigned)f2bf(fj.x) | ((unsigned)f2bf(fj.y) << 16);
        *(unsigned*)&sXj[r][c4 + 2] = (unsigned)f2bf(fj.z) | ((unsigned)f2bf(fj.w) << 16);
    }
    for (int t = tid; t < BT; t += 256) sCol[t] = 0.0f;
    __syncthreads();

    // Preload A fragments for this wave's tile row (reused across 8 tiles).
    // 16-bit A 16x32 layout: lanes 0-15 hold K {ks..ks+7, ks+16..ks+23},
    // lanes 16-31 hold K {ks+8..ks+15, ks+24..ks+31}; M = lane & 15.
    const int tm = wave;
    FragBF a[4];
    {
        const unsigned short* ai = &sXi[tm * 16 + ln][0];
        #pragma unroll
        for (int s = 0; s < 4; ++s) {
            int ks = s * 32;
            a[s].q[0] = *(const u32x4*)(ai + ks + 8 * h);
            a[s].q[1] = *(const u32x4*)(ai + ks + 16 + 8 * h);
        }
    }

    float acc = 0.0f;
    float rowAcc[8];
    #pragma unroll
    for (int r = 0; r < 8; ++r) rowAcc[r] = 0.0f;

    for (int tn = 0; tn < 8; ++tn) {
        v8f c = {};
        const unsigned short* bj = &sXj[tn * 16 + ln][0];
        #pragma unroll
        for (int s = 0; s < 4; ++s) {
            int ks = s * 32;
            FragBF b;
            b.q[0] = *(const u32x4*)(bj + ks + 8 * h);
            b.q[1] = *(const u32x4*)(bj + ks + 16 + 8 * h);
            c = __builtin_amdgcn_wmma_f32_16x16x32_bf16(
                    false, a[s].v, false, b.v, (short)0, c, false, false);
        }

        // L tile in C/D layout: VGPR r -> M = r + 8*h (lanes 0-15 / 16-31), N = ln.
        const float* Lt = L + (size_t)(i0 + tm * 16 + 8 * h) * NN + (j0 + tn * 16 + ln);
        float colp = 0.0f;
        #pragma unroll
        for (int r = 0; r < 8; ++r) {
            float lv = __builtin_nontemporal_load(Lt + (size_t)r * NN);
            acc       += c[r] * lv;
            rowAcc[r] += lv;
            colp      += lv;
        }
        colp += __shfl_xor(colp, 16);
        if (h == 0) atomicAdd(&sCol[tn * 16 + ln], colp);
    }

    // Row sums: reduce across the 16 lanes of each half, one atomic per row.
    #pragma unroll
    for (int r = 0; r < 8; ++r) {
        float v = rowAcc[r];
        v += __shfl_xor(v, 1); v += __shfl_xor(v, 2);
        v += __shfl_xor(v, 4); v += __shfl_xor(v, 8);
        if (ln == 0) atomicAdd(&rowS[i0 + tm * 16 + r + 8 * h], v);
    }

    // Cross: wave reduce -> LDS -> one atomic per workgroup.
    float v = acc;
    v += __shfl_xor(v, 1); v += __shfl_xor(v, 2); v += __shfl_xor(v, 4);
    v += __shfl_xor(v, 8); v += __shfl_xor(v, 16);
    if (lane == 0) sRed[wave] = v;
    __syncthreads();
    if (tid == 0) {
        float s = 0.0f;
        #pragma unroll
        for (int w = 0; w < 8; ++w) s += sRed[w];
        atomicAdd(crossAcc, s);
    }
    if (tid < BT) atomicAdd(&colS[j0 + tid], sCol[tid]);
}

// ---------------------------------------------------------------------------
__global__ __launch_bounds__(256) void sq_kernel(const float* __restrict__ X,
                                                 float* __restrict__ sq)
{
    int row  = blockIdx.x * 8 + (threadIdx.x >> 5);
    int lane = threadIdx.x & 31;
    float4 f = *(const float4*)(X + (size_t)row * DD + lane * 4);
    float s = f.x * f.x + f.y * f.y + f.z * f.z + f.w * f.w;
    s += __shfl_xor(s, 1); s += __shfl_xor(s, 2); s += __shfl_xor(s, 4);
    s += __shfl_xor(s, 8); s += __shfl_xor(s, 16);
    if (lane == 0) sq[row] = s;
}

__global__ __launch_bounds__(256) void ce_kernel(const float* __restrict__ preds,
                                                 const int* __restrict__ targets,
                                                 float* __restrict__ ceAcc)
{
    int row  = blockIdx.x * 8 + (threadIdx.x >> 5);
    int lane = threadIdx.x & 31;
    const float* p = preds + (size_t)row * CC;
    float m = -3.4e38f;
    for (int j = lane; j < CC; j += 32) m = fmaxf(m, p[j]);
    #pragma unroll
    for (int k = 16; k >= 1; k >>= 1) m = fmaxf(m, __shfl_xor(m, k));
    float s = 0.0f;
    for (int j = lane; j < CC; j += 32) s += __expf(p[j] - m);
    #pragma unroll
    for (int k = 16; k >= 1; k >>= 1) s += __shfl_xor(s, k);
    if (lane == 0) {
        int t = targets[row];
        atomicAdd(ceAcc, (m + __logf(s)) - p[t]);   // = -logp[target]
    }
}

__global__ void init_kernel(float* __restrict__ ws)
{
    int i = blockIdx.x * 256 + threadIdx.x;
    if (i < 8 + 2 * NN) ws[i] = 0.0f;
}

__global__ __launch_bounds__(256) void fin_kernel(const float* __restrict__ ws,
                                                  float* __restrict__ out)
{
    const float* rowS = ws + 8;
    const float* colS = ws + 8 + NN;
    const float* sq   = ws + 8 + 2 * NN;
    __shared__ float red[256];
    float s = 0.0f;
    for (int i = threadIdx.x; i < NN; i += 256) s += (rowS[i] + colS[i]) * sq[i];
    red[threadIdx.x] = s;
    __syncthreads();
    for (int k = 128; k >= 1; k >>= 1) {
        if (threadIdx.x < k) red[threadIdx.x] += red[threadIdx.x + k];
        __syncthreads();
    }
    if (threadIdx.x == 0)
        out[0] = ws[1] * (1.0f / NN) + ALPHA_C * (red[0] - 2.0f * ws[0]);
}

// ---------------------------------------------------------------------------
extern "C" void kernel_launch(void* const* d_in, const int* in_sizes, int n_in,
                              void* d_out, int out_size, void* d_ws, size_t ws_size,
                              hipStream_t stream)
{
    const float* preds   = (const float*)d_in[0];
    const int*   targets = (const int*)d_in[1];
    const float* Lmat    = (const float*)d_in[2];
    const float* X       = (const float*)d_in[3];
    float* out = (float*)d_out;
    float* ws  = (float*)d_ws;   // [0]=cross, [1]=ce, +8: row[N], +8+N: col[N], +8+2N: sq[N]

    init_kernel<<<(8 + 2 * NN + 255) / 256, 256, 0, stream>>>(ws);
    sq_kernel<<<NN / 8, 256, 0, stream>>>(X, ws + 8 + 2 * NN);
    ce_kernel<<<NN / 8, 256, 0, stream>>>(preds, targets, ws + 1);
    dim3 grid(NN / BT, NN / BT);
    cross_kernel<<<grid, 256, 0, stream>>>(Lmat, X, ws + 0, ws + 8, ws + 8 + NN);
    fin_kernel<<<1, 256, 0, stream>>>(ws, out);
}
